// Attention_9242769621473
// MI455X (gfx1250) — matrix operand order repaired and checked by reading, not compiled, not run on hardware
//
#include <hip/hip_runtime.h>

// ---------------------------------------------------------------------------
// CDNA5 (gfx1250, wave32) fused attention:
//   3x projection GEMM (f32 -> bf16 WMMA -> bf16 ws), double-buffered LDS
//   flash attention    (bf16 WMMA, f32 online softmax, WMMA row-sums,
//                       DPP16 row-max, pipelined TDM K-staging)
//   output GEMM        (bf16 -> f32 out)
// ---------------------------------------------------------------------------

typedef __attribute__((ext_vector_type(16))) __bf16 v16bf;
typedef __attribute__((ext_vector_type(8)))  float  v8f;
typedef __attribute__((ext_vector_type(4)))  unsigned int v4u;
typedef __attribute__((ext_vector_type(8)))  int  v8i;
typedef __attribute__((ext_vector_type(4)))  int  v4i;

union Frag16 {
  v16bf v;
  unsigned int u[8];
};

__device__ __forceinline__ unsigned short f2bf(float f) {
  unsigned int u = __builtin_bit_cast(unsigned int, f);
  u += 0x7FFFu + ((u >> 16) & 1u);   // round-to-nearest-even
  return (unsigned short)(u >> 16);
}
__device__ __forceinline__ unsigned int pack2bf(float lo, float hi) {
  return (unsigned int)f2bf(lo) | ((unsigned int)f2bf(hi) << 16);
}

// 16-lane max reduction on VALU via DPP16 (no LDS, no dep-counter waits).
// After xor1+xor2 each quad is uniform, so row_half_mirror (xor7) acts as
// xor4 and row_mirror (xor15) acts as xor8.
template <int CTRL>
__device__ __forceinline__ float max_dpp(float x) {
#if __has_builtin(__builtin_amdgcn_update_dpp)
  int s = __builtin_bit_cast(int, x);
  int p = __builtin_amdgcn_update_dpp(s, s, CTRL, 0xF, 0xF, true);
  return fmaxf(x, __builtin_bit_cast(float, p));
#else
  return x;
#endif
}
__device__ __forceinline__ float rowmax16(float x) {
#if __has_builtin(__builtin_amdgcn_update_dpp)
  x = max_dpp<0xB1>(x);   // quad_perm [1,0,3,2] : xor 1
  x = max_dpp<0x4E>(x);   // quad_perm [2,3,0,1] : xor 2
  x = max_dpp<0x141>(x);  // row_half_mirror     : xor 4 on quad-uniform data
  x = max_dpp<0x140>(x);  // row_mirror          : xor 8 on oct-uniform data
#else
#pragma unroll
  for (int m = 1; m <= 8; m <<= 1) x = fmaxf(x, __shfl_xor(x, m, 32));
#endif
  return x;
}

namespace cfg {
constexpr int B = 4, N = 2048, E = 1024, H = 16, D = 1024, HD = 64;
constexpr int M = B * N;  // 8192 GEMM rows
}

#if __has_builtin(__builtin_amdgcn_tensor_load_to_lds)
#define HAVE_TDM 1
#else
#define HAVE_TDM 0
#endif

#if HAVE_TDM
// TDM: 2D tile 64(d) x 32(keys), 2-byte elems, tensor row stride D halves.
// LDS pad: +1 DWORD after every 32 DWORDs -> LDS row stride 66 halves.
__device__ __forceinline__ void tdm_load_tile_k(const unsigned short* gsrc,
                                                unsigned short* ldst) {
  const unsigned long long ga = (unsigned long long)(const void*)gsrc;
  const unsigned int lds = (unsigned int)(unsigned long long)(void*)ldst;
  v4u g0 = {1u,                                   // count=1 (valid D#)
            lds,                                  // lds_addr
            (unsigned int)(ga & 0xFFFFFFFFull),   // global_addr[31:0]
            (unsigned int)((ga >> 32) & 0x1FFFFFFull) | (2u << 30)};  // type=2
  v8i g1 = {(int)((1u << 16) | (1u << 20) | (4u << 22)),  // 2B, pad_en, 32dw/1dw
            (int)((unsigned)cfg::D << 16),                // tensor_dim0 lo16
            (int)(((unsigned)cfg::D >> 16) | ((unsigned)cfg::N << 16)),
            (int)((((unsigned)cfg::N >> 16) & 0xFFFFu) | (64u << 16)),  // tile0=64
            (int)32u,                                     // tile_dim1=32
            (int)(unsigned)cfg::D,                        // dim0_stride lo32
            0, 0};
  v4i gz = {0, 0, 0, 0};
#if __clang_major__ >= 23
  v8i gz8 = {0, 0, 0, 0, 0, 0, 0, 0};
  __builtin_amdgcn_tensor_load_to_lds(g0, g1, gz, gz, gz8, 0);
#else
  __builtin_amdgcn_tensor_load_to_lds(g0, g1, gz, gz, 0);
#endif
}
__device__ __forceinline__ void tdm_wait() {
#if __has_builtin(__builtin_amdgcn_s_wait_tensorcnt)
  __builtin_amdgcn_s_wait_tensorcnt(0);
#endif
}
#endif

// ===========================================================================
// GEMM: C[M,Nc] = A[M,K] * W[K,Nc] + bias    (bf16 WMMA, f32 accumulate)
// Block = 256 threads = 8 waves; tile 128(M) x 128(N) x 32(K).
// Wave w owns a 32x64 output tile (2x4 WMMA fragments).
// Double-buffered LDS: one barrier per K-step, global loads overlapped.
// ===========================================================================
template <bool A_BF16, bool OUT_BF16>
__global__ void __launch_bounds__(256)
gemm_wmma(const void* __restrict__ Ap, const float* __restrict__ W,
          const float* __restrict__ bias, void* __restrict__ Cp,
          int M, int K, int Nc) {
  constexpr int TM = 128, TN = 128, TK = 32;
  constexpr int LDA = TK + 2, LDB = TK + 2;   // padded, bank-conflict free
  __shared__ unsigned short As[2][TM * LDA];  // As[buf][m][k]
  __shared__ unsigned short Bs[2][TN * LDB];  // Bs[buf][n][k] (transposed)

  const int tid  = threadIdx.x;
  const int lane = tid & 31, wave = tid >> 5;
  const int half = lane >> 4, l16 = lane & 15;
  const int m0 = blockIdx.y * TM, n0 = blockIdx.x * TN;
  const int wm = (wave >> 1) * 32, wn = (wave & 1) * 64;

  const int arow = tid >> 1, acol = (tid & 1) * 16;  // 16 A elements / thread
  const int brow = tid >> 3, bcol = (tid & 7) * 16;  // 16 W elements / thread

  v8f acc[2][4] = {};

  uint4  aregb[2];
  float4 aregf[4];
  float4 wreg[4];

  auto load_regs = [&](int k0) {
    if constexpr (A_BF16) {
      const unsigned short* ag =
          (const unsigned short*)Ap + (size_t)(m0 + arow) * K + k0 + acol;
      const uint4* ag4 = (const uint4*)ag;
      aregb[0] = ag4[0];
      aregb[1] = ag4[1];
      if (k0 + TK < K) __builtin_prefetch(ag + TK, 0, 1);
    } else {
      const float* af = (const float*)Ap + (size_t)(m0 + arow) * K + k0 + acol;
      const float4* af4 = (const float4*)af;
#pragma unroll
      for (int j = 0; j < 4; ++j) aregf[j] = af4[j];
      if (k0 + TK < K) __builtin_prefetch(af + TK, 0, 1);
    }
    const float* wf = W + (size_t)(k0 + brow) * Nc + n0 + bcol;
    const float4* wf4 = (const float4*)wf;
#pragma unroll
    for (int g = 0; g < 4; ++g) wreg[g] = wf4[g];
    if (k0 + TK < K) __builtin_prefetch(wf + (size_t)TK * Nc, 0, 1);
  };

  auto store_regs = [&](int buf) {
    unsigned int* adst = (unsigned int*)&As[buf][arow * LDA + acol];
    if constexpr (A_BF16) {
      adst[0] = aregb[0].x; adst[1] = aregb[0].y;
      adst[2] = aregb[0].z; adst[3] = aregb[0].w;
      adst[4] = aregb[1].x; adst[5] = aregb[1].y;
      adst[6] = aregb[1].z; adst[7] = aregb[1].w;
    } else {
#pragma unroll
      for (int j = 0; j < 4; ++j) {
        adst[2 * j + 0] = pack2bf(aregf[j].x, aregf[j].y);
        adst[2 * j + 1] = pack2bf(aregf[j].z, aregf[j].w);
      }
    }
#pragma unroll
    for (int g = 0; g < 4; ++g) {
      Bs[buf][(bcol + g * 4 + 0) * LDB + brow] = f2bf(wreg[g].x);
      Bs[buf][(bcol + g * 4 + 1) * LDB + brow] = f2bf(wreg[g].y);
      Bs[buf][(bcol + g * 4 + 2) * LDB + brow] = f2bf(wreg[g].z);
      Bs[buf][(bcol + g * 4 + 3) * LDB + brow] = f2bf(wreg[g].w);
    }
  };

  // prologue: stage k-slab 0
  load_regs(0);
  store_regs(0);
  __syncthreads();

  const int TSTEPS = K / TK;
  for (int t = 0; t < TSTEPS; ++t) {
    const int cur = t & 1, nxt = cur ^ 1;
    const bool more = (t + 1) < TSTEPS;
    if (more) load_regs((t + 1) * TK);   // overlap next slab with compute

    Frag16 af2[2], bf4[4];
#pragma unroll
    for (int vv = 0; vv < 8; ++vv) {
      const int ka = ((vv & 3) * 2) + ((vv >> 2) * 16) + half * 8;  // A layout
      const int kb = ((vv & 3) * 2) + ((vv >> 2) * 16) + half * 8;  // B layout
      af2[0].u[vv] = *(const unsigned int*)&As[cur][(wm + l16) * LDA + ka];
      af2[1].u[vv] = *(const unsigned int*)&As[cur][(wm + 16 + l16) * LDA + ka];
#pragma unroll
      for (int j = 0; j < 4; ++j)
        bf4[j].u[vv] =
            *(const unsigned int*)&Bs[cur][(wn + j * 16 + l16) * LDB + kb];
    }
#pragma unroll
    for (int i = 0; i < 2; ++i)
#pragma unroll
      for (int j = 0; j < 4; ++j)
        acc[i][j] = __builtin_amdgcn_wmma_f32_16x16x32_bf16(
            false, af2[i].v, false, bf4[j].v, (short)0, acc[i][j], false, false);

    if (more) store_regs(nxt);
    __syncthreads();
  }

  // ---- epilogue: +bias, write bf16 (ws) or f32 (final output) ----
#pragma unroll
  for (int i = 0; i < 2; ++i) {
#pragma unroll
    for (int j = 0; j < 4; ++j) {
      const int gn = n0 + wn + j * 16 + l16;
      const float bb = bias[gn];
      const int gmBase = m0 + wm + i * 16 + half * 8;
#pragma unroll
      for (int r = 0; r < 8; ++r) {
        float val = acc[i][j][r] + bb;
        size_t idx = (size_t)(gmBase + r) * Nc + gn;
        if (OUT_BF16) ((unsigned short*)Cp)[idx] = f2bf(val);
        else          ((float*)Cp)[idx] = val;
      }
    }
  }
}

// ===========================================================================
// Flash attention over bf16 Q/K/V in ws layout [B, N, H*HD].
// Block = 256 threads = 8 waves; each wave owns 32 q-rows x 64 d.
// Pipelined: TDM K-load + V global loads for chunk t+1 issued before the
// compute of chunk t; one barrier per chunk. Row-sum via WMMA x ones,
// row-max via DPP16.
// ===========================================================================
__global__ void __launch_bounds__(256)
flash_attn(const unsigned short* __restrict__ Qp,
           const unsigned short* __restrict__ Kp,
           const unsigned short* __restrict__ Vp,
           unsigned short* __restrict__ Op) {
  using namespace cfg;
  constexpr int LKV = HD + 2;   // Ks[key][d]   row stride (halves)
  constexpr int LVT = 32 + 2;   // Vs[d][key]   row stride
  constexpr int LPS = 32 + 2;   // Ps[m][key]   row stride (per-wave)
  constexpr int TSTEPS = N / 32;
  __shared__ unsigned short Ks[2][32 * LKV];
  __shared__ unsigned short Vs[2][HD * LVT];
  __shared__ unsigned short Ps[8 * 32 * LPS];

  const int tid  = threadIdx.x;
  const int lane = tid & 31, wave = tid >> 5;
  const int half = lane >> 4, l16 = lane & 15;
  const int bh = blockIdx.y;
  const int b = bh / H, h = bh % H;
  const int qbase = blockIdx.x * 256 + wave * 32;

  const unsigned short* Q  = Qp + (size_t)b * N * D + h * HD;
  const unsigned short* Kg = Kp + (size_t)b * N * D + h * HD;
  const unsigned short* Vg = Vp + (size_t)b * N * D + h * HD;
  unsigned short*       Og = Op + (size_t)b * N * D + h * HD;

  // Q fragments [msub][dchunk], gathered straight from global (loaded once)
  Frag16 qf[2][2];
#pragma unroll
  for (int i = 0; i < 2; ++i) {
    const unsigned short* qrow = Q + (size_t)(qbase + i * 16 + l16) * D;
#pragma unroll
    for (int c = 0; c < 2; ++c)
#pragma unroll
      for (int vv = 0; vv < 8; ++vv) {
        const int k = c * 32 + ((vv & 3) * 2) + ((vv >> 2) * 16) + half * 8;
        qf[i][c].u[vv] = *(const unsigned int*)(qrow + k);
      }
  }

  Frag16 ones;                       // bf16 1.0 replicated: P x ones = rowsum
#pragma unroll
  for (int vv = 0; vv < 8; ++vv) ones.u[vv] = 0x3F803F80u;

  v8f o[2][4] = {};
  v8f lsum[2] = {};
  float mrun[2][8];
#pragma unroll
  for (int i = 0; i < 2; ++i)
#pragma unroll
    for (int r = 0; r < 8; ++r) mrun[i][r] = -1e30f;

  const float cexp = 0.125f * 1.44269504088896340736f;  // 1/sqrt(64) * log2(e)
  unsigned short* Pw = Ps + wave * 32 * LPS;

  const int krow = tid >> 3;        // 0..31 : key row within chunk
  const int dgrp = (tid & 7) * 8;   // 0..56 : 8 d-values per thread

  // ---------------- prologue: stage chunk 0 ----------------
#if HAVE_TDM
  if (wave == 0) tdm_load_tile_k(Kg, &Ks[0][0]);
#else
  {
    uint4 k4 = *(const uint4*)(Kg + (size_t)krow * D + dgrp);
    unsigned int* kd = (unsigned int*)&Ks[0][krow * LKV + dgrp];
    kd[0] = k4.x; kd[1] = k4.y; kd[2] = k4.z; kd[3] = k4.w;
  }
#endif
  {
    uint4 v4 = *(const uint4*)(Vg + (size_t)krow * D + dgrp);
    unsigned int vu[4] = {v4.x, v4.y, v4.z, v4.w};
#pragma unroll
    for (int j = 0; j < 8; ++j)
      Vs[0][(dgrp + j) * LVT + krow] =
          (unsigned short)(vu[j >> 1] >> ((j & 1) * 16));
  }
#if HAVE_TDM
  if (wave == 0) tdm_wait();
#endif
  __syncthreads();

  for (int t = 0; t < TSTEPS; ++t) {
    const int cur = t & 1, nxt = cur ^ 1;
    const bool more = (t + 1) < TSTEPS;
    const unsigned short* KsC = &Ks[cur][0];
    const unsigned short* VsC = &Vs[cur][0];

    // ---- issue chunk t+1 loads; latency hidden behind this chunk's math ----
    uint4 vnext = {0, 0, 0, 0};
#if HAVE_TDM
    if (more && wave == 0)
      tdm_load_tile_k(Kg + (size_t)(t + 1) * 32 * D, &Ks[nxt][0]);
#else
    uint4 knext = {0, 0, 0, 0};
    if (more)
      knext = *(const uint4*)(Kg + (size_t)((t + 1) * 32 + krow) * D + dgrp);
#endif
    if (more)
      vnext = *(const uint4*)(Vg + (size_t)((t + 1) * 32 + krow) * D + dgrp);

    // ---- S = Q * K^T : 32(q) x 32(keys) per wave, HD in 2 k-chunks ----
    v8f s[2][2] = {};
#pragma unroll
    for (int c = 0; c < 2; ++c) {
#pragma unroll
      for (int j = 0; j < 2; ++j) {
        Frag16 kf;
#pragma unroll
        for (int vv = 0; vv < 8; ++vv)
          kf.u[vv] = *(const unsigned int*)
              &KsC[(j * 16 + l16) * LKV + c * 32 + ((vv & 3) * 2) + ((vv >> 2) * 16) + half * 8];
#pragma unroll
        for (int i = 0; i < 2; ++i)
          s[i][j] = __builtin_amdgcn_wmma_f32_16x16x32_bf16(
              false, qf[i][c].v, false, kf.v, (short)0, s[i][j], false, false);
      }
    }

    // ---- row max on VALU (DPP16 tree) ----
    float rmax[2][8];
#pragma unroll
    for (int i = 0; i < 2; ++i)
#pragma unroll
      for (int r = 0; r < 8; ++r)
        rmax[i][r] = rowmax16(fmaxf(s[i][0][r], s[i][1][r]));

    // ---- online-softmax rescale; P -> bf16 via per-wave LDS transpose ----
#pragma unroll
    for (int i = 0; i < 2; ++i) {
      float mnew[8], corr[8];
#pragma unroll
      for (int r = 0; r < 8; ++r) {
        mnew[r] = fmaxf(mrun[i][r], rmax[i][r]);
        corr[r] = exp2f((mrun[i][r] - mnew[r]) * cexp);
        mrun[i][r] = mnew[r];
        lsum[i][r] *= corr[r];
      }
#pragma unroll
      for (int jd = 0; jd < 4; ++jd)
#pragma unroll
        for (int r = 0; r < 8; ++r) o[i][jd][r] *= corr[r];
#pragma unroll
      for (int j = 0; j < 2; ++j)
#pragma unroll
        for (int r = 0; r < 8; ++r) {
          float p = exp2f((s[i][j][r] - mnew[r]) * cexp);
          Pw[(i * 16 + r + half * 8) * LPS + j * 16 + l16] = f2bf(p);
        }
    }

    // ---- O += P*V ; l += P*ones  (row-sum on the matrix engine) ----
#pragma unroll
    for (int i = 0; i < 2; ++i) {
      Frag16 pa;
#pragma unroll
      for (int vv = 0; vv < 8; ++vv) {
        const int k = ((vv & 3) * 2) + ((vv >> 2) * 16) + half * 8;
        pa.u[vv] = *(const unsigned int*)&Pw[(i * 16 + l16) * LPS + k];
      }
      lsum[i] = __builtin_amdgcn_wmma_f32_16x16x32_bf16(
          false, pa.v, false, ones.v, (short)0, lsum[i], false, false);
#pragma unroll
      for (int jd = 0; jd < 4; ++jd) {
        Frag16 vf;
#pragma unroll
        for (int vv = 0; vv < 8; ++vv)
          vf.u[vv] = *(const unsigned int*)
              &VsC[(jd * 16 + l16) * LVT + ((vv & 3) * 2) + ((vv >> 2) * 16) + half * 8];
        o[i][jd] = __builtin_amdgcn_wmma_f32_16x16x32_bf16(
            false, pa.v, false, vf.v, (short)0, o[i][jd], false, false);
      }
    }

    // ---- stage chunk t+1 into the other buffer; single barrier ----
    if (more) {
#if !HAVE_TDM
      unsigned int* kd = (unsigned int*)&Ks[nxt][krow * LKV + dgrp];
      kd[0] = knext.x; kd[1] = knext.y; kd[2] = knext.z; kd[3] = knext.w;
#endif
      unsigned int vu[4] = {vnext.x, vnext.y, vnext.z, vnext.w};
#pragma unroll
      for (int j = 0; j < 8; ++j)
        Vs[nxt][(dgrp + j) * LVT + krow] =
            (unsigned short)(vu[j >> 1] >> ((j & 1) * 16));
    }
#if HAVE_TDM
    if (more && wave == 0) tdm_wait();
#endif
    __syncthreads();
  }

  // ---- finalize: O /= l, store bf16 into ws (heads recombined layout) ----
#pragma unroll
  for (int i = 0; i < 2; ++i)
#pragma unroll
    for (int jd = 0; jd < 4; ++jd)
#pragma unroll
      for (int r = 0; r < 8; ++r) {
        float val = o[i][jd][r] / lsum[i][r];
        Og[(size_t)(qbase + i * 16 + r + half * 8) * D + jd * 16 + l16] =
            f2bf(val);
      }
}

// ===========================================================================
extern "C" void kernel_launch(void* const* d_in, const int* in_sizes, int n_in,
                              void* d_out, int out_size, void* d_ws,
                              size_t ws_size, hipStream_t stream) {
  (void)in_sizes; (void)n_in; (void)out_size; (void)ws_size;
  using namespace cfg;

  const float* q  = (const float*)d_in[0];
  const float* k  = (const float*)d_in[1];
  const float* v  = (const float*)d_in[2];
  const float* Wq = (const float*)d_in[3];
  const float* bq = (const float*)d_in[4];
  const float* Wk = (const float*)d_in[5];
  const float* bk = (const float*)d_in[6];
  const float* Wv = (const float*)d_in[7];
  const float* bv = (const float*)d_in[8];
  const float* Wo = (const float*)d_in[9];
  const float* bo = (const float*)d_in[10];

  // ws: QP | KP | VP | O  as bf16, each M*D halves (16 MiB) -> 64 MiB total
  unsigned short* qp = (unsigned short*)d_ws;
  unsigned short* kp = qp + (size_t)M * D;
  unsigned short* vp = kp + (size_t)M * D;
  unsigned short* op = vp + (size_t)M * D;

  dim3 blk(256);
  dim3 gg(D / 128, M / 128);  // 8 x 64 blocks
  gemm_wmma<false, true><<<gg, blk, 0, stream>>>(q, Wq, bq, qp, M, E, D);
  gemm_wmma<false, true><<<gg, blk, 0, stream>>>(k, Wk, bk, kp, M, E, D);
  gemm_wmma<false, true><<<gg, blk, 0, stream>>>(v, Wv, bv, vp, M, E, D);

  dim3 gf(N / 256, B * H);    // 8 x 64 blocks
  flash_attn<<<gf, blk, 0, stream>>>(qp, kp, vp, op);

  dim3 go(E / 128, M / 128);
  gemm_wmma<true, false><<<go, blk, 0, stream>>>(op, Wo, bo, d_out, M, D, E);
}
